// TransformerEncoder_87119116632418
// MI455X (gfx1250) — matrix operand order repaired and checked by reading, not compiled, not run on hardware
//
#include <hip/hip_runtime.h>

// ---------------------------------------------------------------------------
// MI455X (gfx1250, wave32) fused transformer-encoder block.
// All GEMMs run on v_wmma_f32_16x16x32_bf16 (f32 accumulate); epilogues
// (bias / relu / residual / LayerNorm) fused in f32. Attention collapses to a
// closed form because scores are mask-only (2 distinct ctx rows per batch).
//
// Round-1 fix: A operands are stored in LDS pre-swizzled into the WMMA lane
// layout (one contiguous 32B load per lane), and the K-loop is outermost with
// per-wave persistent accumulators -> no register spills, no VALU shuffles.
// ---------------------------------------------------------------------------

typedef __attribute__((ext_vector_type(16))) __bf16 v16bf;
typedef __attribute__((ext_vector_type(8)))  float  v8f;

#define BT    256
#define WAVES 8

constexpr int Bb    = 512;
constexpr int Ss    = 100;
constexpr int Dd    = 512;
constexpr int DFFc  = 256;
constexpr int HIDc  = 230;
constexpr int Nrows = Bb * Ss;   // 51200

__device__ __forceinline__ unsigned short f2bf(float f) {
  unsigned u = __builtin_bit_cast(unsigned, f);
  u += 0x7FFFu + ((u >> 16) & 1u);               // round-to-nearest-even
  return (unsigned short)(u >> 16);
}

__device__ __forceinline__ float wave_sum(float x) {
  #pragma unroll
  for (int m = 16; m > 0; m >>= 1) x += __shfl_xor(x, m, 32);
  return x;
}

// ---------------- conversion kernels ----------------
__global__ void k_cvt_bf16(const float* __restrict__ s, unsigned short* __restrict__ d, int n) {
  int i = blockIdx.x * blockDim.x + threadIdx.x;
  if (i < n) d[i] = f2bf(s[i]);
}

// pad mlpf_w [230,512] -> [240,512] (zeros) so it tiles as 15 WMMA n-tiles
__global__ void k_pad_mlpf(const float* __restrict__ s, unsigned short* __restrict__ d) {
  int i = blockIdx.x * blockDim.x + threadIdx.x;
  if (i < 240 * Dd) {
    int r = i >> 9;
    d[i] = (r < HIDc) ? f2bf(s[i]) : (unsigned short)0;
  }
}

// ---------------- cooperative A load: global row-major -> LDS WMMA-swizzled -
// A tile is 16 x K bf16 (16 consecutive rows, contiguous in global).
// A operand k order used here (16-bit, 16x32 per K-step; B uses the same):
//   lane L (hi=L>>4, m=L&15) holds h[j]   = A[m][kt*32 + hi*16 + j]     j=0..15
// We store fragment kt, lane L as 16 contiguous bf16 at (kt*32 + L)*16 so the
// compute loop does a single 32-byte LDS load per lane (no shuffles).
template<int K>
__device__ __forceinline__ void load_A_swizzled(const unsigned short* __restrict__ A,
                                                int row0, unsigned short* __restrict__ Alds) {
  const int tid = threadIdx.x;
  const uint2* src = (const uint2*)(A + (size_t)row0 * K);   // 8 bf16 per chunk
  constexpr int CPR = K / 8;          // chunks per row
  for (int i = tid; i < 16 * CPR; i += BT) {
    const int m  = i / CPR;
    const int c  = i % CPR;           // global k = c*8
    uint2 d = src[i];
    const int kt   = c >> 2;                        // K-step
    const int lane = ((c & 2) ? 16 : 0) + m;        // hi half from k0&16
    const int jb   = (c & 1) ? 8 : 0;               // j-base from k0&8
    *(uint2*)(Alds + (kt * 32 + lane) * 16 + jb) = d;
  }
}

// ---------------- GEMM, direct epilogue (bias / relu / f32-or-bf16 out) -----
// W is [n_out][K] bf16 row-major == WMMA B operand in K-major columns
// (lane nlo -> N, 16 contiguous K per half-wave).
template<int K, int NTILES, int ACT, int OUTBF, int BIAS>
__global__ void k_gemm(const unsigned short* __restrict__ A,
                       const unsigned short* __restrict__ W,
                       const float* __restrict__ bias,
                       void* __restrict__ out, int ncols) {
  constexpr int NTPW = (NTILES + WAVES - 1) / WAVES;
  __shared__ unsigned short Alds[16 * K];
  const int tid  = threadIdx.x;
  const int row0 = blockIdx.x * 16;
  load_A_swizzled<K>(A, row0, Alds);
  __syncthreads();

  const int wave = tid >> 5;
  const int lane = tid & 31;
  const int nlo  = lane & 15;
  const int hi   = lane >> 4;

  v8f acc[NTPW] = {};
  const unsigned short* wbase = W + (size_t)(wave * 16 + nlo) * K + hi * 16;

  #pragma unroll 4
  for (int kk = 0; kk < K; kk += 32) {
    v16bf a = *(const v16bf*)(Alds + ((kk >> 5) * 32 + lane) * 16);
    #pragma unroll
    for (int t = 0; t < NTPW; ++t) {
      if (NTILES % WAVES == 0 || wave + t * WAVES < NTILES) {   // wave-uniform
        v16bf b = *(const v16bf*)(wbase + (size_t)t * WAVES * 16 * K + kk);
        acc[t] = __builtin_amdgcn_wmma_f32_16x16x32_bf16(false, a, false, b,
                                                         (short)0, acc[t], false, false);
      }
    }
  }

  #pragma unroll
  for (int t = 0; t < NTPW; ++t) {
    const int tn = wave + t * WAVES;
    if (NTILES % WAVES != 0 && tn >= NTILES) continue;
    const int col = tn * 16 + nlo;
    if (col < ncols) {
      #pragma unroll
      for (int j = 0; j < 8; ++j) {
        const int m = hi * 8 + j;
        float v = acc[t][j];
        if (BIAS) v += bias[col];
        if (ACT)  v = v > 0.f ? v : 0.f;
        size_t o = (size_t)(row0 + m) * ncols + col;
        if (OUTBF) ((unsigned short*)out)[o] = f2bf(v);
        else       ((float*)out)[o] = v;
      }
    }
  }
}

// ---------------- GEMM + custom LayerNorm epilogue ----------------
// MODE 0: x = acc + bias + resid; y = LN(x); store bf16(y + post_add)
// MODE 1: x = acc + bias*bscale;  y = LN(x); store f32(y) and bf16(y)
// MODE 2: x = acc + bias + resid; y = LN(x); store bf16(y)
// Custom LN: ddof=1 std, eps added to std (matches reference _ln_custom).
template<int K, int MODE>
__global__ void k_gemm_ln(const unsigned short* __restrict__ A,
                          const unsigned short* __restrict__ W,
                          const float* __restrict__ bias, float bscale,
                          const float* __restrict__ resid,
                          const float* __restrict__ gamma,
                          const float* __restrict__ beta,
                          const float* __restrict__ post_add,
                          float* __restrict__ outf,
                          unsigned short* __restrict__ outb) {
  constexpr int NTPW = (Dd / 16) / WAVES;   // 4
  constexpr int CST  = Dd + 4;              // LDS pad vs bank conflicts
  __shared__ unsigned short Alds[16 * K];
  __shared__ float Clds[16 * CST];
  const int tid  = threadIdx.x;
  const int row0 = blockIdx.x * 16;
  load_A_swizzled<K>(A, row0, Alds);
  __syncthreads();

  const int wave = tid >> 5;
  const int lane = tid & 31;
  const int nlo  = lane & 15;
  const int hi   = lane >> 4;

  v8f acc[NTPW] = {};
  const unsigned short* wbase = W + (size_t)(wave * 16 + nlo) * K + hi * 16;

  #pragma unroll 4
  for (int kk = 0; kk < K; kk += 32) {
    v16bf a = *(const v16bf*)(Alds + ((kk >> 5) * 32 + lane) * 16);
    #pragma unroll
    for (int t = 0; t < NTPW; ++t) {
      v16bf b = *(const v16bf*)(wbase + (size_t)t * WAVES * 16 * K + kk);
      acc[t] = __builtin_amdgcn_wmma_f32_16x16x32_bf16(false, a, false, b,
                                                       (short)0, acc[t], false, false);
    }
  }

  #pragma unroll
  for (int t = 0; t < NTPW; ++t) {
    const int col = (wave + t * WAVES) * 16 + nlo;
    #pragma unroll
    for (int j = 0; j < 8; ++j) {
      const int m = hi * 8 + j;
      float v = acc[t][j] + bias[col] * bscale;
      if (MODE != 1) v += resid[(size_t)(row0 + m) * Dd + col];
      Clds[m * CST + col] = v;
    }
  }
  __syncthreads();

  for (int r = wave * 2; r < wave * 2 + 2; ++r) {
    float s = 0.f, s2 = 0.f;
    #pragma unroll
    for (int c = lane; c < Dd; c += 32) {
      float x = Clds[r * CST + c];
      s += x; s2 += x * x;
    }
    s = wave_sum(s); s2 = wave_sum(s2);
    const float mean = s * (1.0f / Dd);
    float varu = (s2 - (float)Dd * mean * mean) * (1.0f / (Dd - 1));
    varu = varu > 0.f ? varu : 0.f;
    const float inv = 1.0f / (sqrtf(varu) + 1e-6f);
    const size_t ro = (size_t)(row0 + r) * Dd;
    #pragma unroll
    for (int c = lane; c < Dd; c += 32) {
      float x = Clds[r * CST + c];
      float y = gamma[c] * (x - mean) * inv + beta[c];
      if (MODE == 0)      outb[ro + c] = f2bf(y + post_add[ro + c]);
      else if (MODE == 1) { outf[ro + c] = y; outb[ro + c] = f2bf(y); }
      else                outb[ro + c] = f2bf(y);
    }
  }
}

// ---------------- mask-only attention, closed form ----------------
// p rows take 2 forms per batch -> ctx has only 2 distinct rows per batch.
__global__ void k_ctx(const float* __restrict__ v, const int* __restrict__ mask,
                      unsigned short* __restrict__ ctx2) {
  const int b   = blockIdx.x;
  const int tid = threadIdx.x;
  __shared__ int ms[Ss];
  if (tid < Ss) ms[tid] = mask[b * Ss + tid];
  __syncthreads();
  int n1 = 0;
  #pragma unroll 4
  for (int t = 0; t < Ss; ++t) n1 += ms[t];
  const float e1 = 2.718281828459045f;   // exp(1)
  const float e0 = 1.0f;                 // exp(1e-8) rounds to 1.0f
  const float Z  = (float)n1 * e1 + (float)(Ss - n1) * e0;
  for (int d = tid; d < Dd; d += BT) {
    float sa = 0.f, sm = 0.f;
    for (int t = 0; t < Ss; ++t) {
      float x = v[((size_t)b * Ss + t) * Dd + d];
      sa += x;
      sm += ms[t] ? x : 0.f;
    }
    float ctxU = (e1 * sm + e0 * (sa - sm)) / Z;   // unmasked query row
    float ctxM = sa * (1.0f / Ss);                 // masked row -> uniform
    ctx2[((size_t)b * 2 + 0) * Dd + d] = f2bf(ctxM);
    ctx2[((size_t)b * 2 + 1) * Dd + d] = f2bf(ctxU);
  }
}

// ---------------- mha = LN_torch(select(g2) + fc_b + inp) ----------------
__global__ void k_mha_ln(const float* __restrict__ g2, const int* __restrict__ mask,
                         const float* __restrict__ fc_b, const float* __restrict__ inp,
                         const float* __restrict__ lng, const float* __restrict__ lnb,
                         float* __restrict__ mhaf, unsigned short* __restrict__ mhab) {
  const int row = blockIdx.x;            // == b*S + s
  const int b   = row / Ss;
  const int sel = (mask[row] != 0) ? 1 : 0;
  const float* gr = g2 + ((size_t)b * 2 + sel) * Dd;
  const size_t ro = (size_t)row * Dd;
  const int tid = threadIdx.x;
  float x0 = gr[tid]       + fc_b[tid]       + inp[ro + tid];
  float x1 = gr[tid + 256] + fc_b[tid + 256] + inp[ro + tid + 256];
  float s  = x0 + x1, s2 = x0 * x0 + x1 * x1;
  s = wave_sum(s); s2 = wave_sum(s2);
  __shared__ float rb[16];
  const int wave = tid >> 5, lane = tid & 31;
  if (lane == 0) { rb[wave] = s; rb[8 + wave] = s2; }
  __syncthreads();
  float S = 0.f, S2 = 0.f;
  #pragma unroll
  for (int w = 0; w < 8; ++w) { S += rb[w]; S2 += rb[8 + w]; }
  const float mean = S * (1.0f / Dd);
  const float var  = S2 * (1.0f / Dd) - mean * mean;   // biased (torch LN)
  const float inv  = rsqrtf(var + 1e-5f);
  float y0 = lng[tid]       * (x0 - mean) * inv + lnb[tid];
  float y1 = lng[tid + 256] * (x1 - mean) * inv + lnb[tid + 256];
  mhaf[ro + tid] = y0;        mhaf[ro + tid + 256] = y1;
  mhab[ro + tid] = f2bf(y0);  mhab[ro + tid + 256] = f2bf(y1);
}

// ---------------------------------------------------------------------------
extern "C" void kernel_launch(void* const* d_in, const int* in_sizes, int n_in,
                              void* d_out, int out_size, void* d_ws, size_t ws_size,
                              hipStream_t stream) {
  (void)in_sizes; (void)n_in; (void)out_size; (void)ws_size;
  const float* inp    = (const float*)d_in[0];
  const int*   mask   = (const int*)d_in[1];
  /* d_in[2] = gate: dead computation in reference */
  const float* wv     = (const float*)d_in[3];
  const float* bv     = (const float*)d_in[4];
  const float* fc_w   = (const float*)d_in[5];
  const float* fc_b   = (const float*)d_in[6];
  const float* lnm_g  = (const float*)d_in[7];
  const float* lnm_b  = (const float*)d_in[8];
  const float* w1     = (const float*)d_in[9];
  const float* b1     = (const float*)d_in[10];
  const float* w2     = (const float*)d_in[11];
  const float* b2     = (const float*)d_in[12];
  const float* lpc_w  = (const float*)d_in[13];
  const float* lpc_b  = (const float*)d_in[14];
  const float* lnc_g  = (const float*)d_in[15];
  const float* lnc_b  = (const float*)d_in[16];
  const float* pwf_w  = (const float*)d_in[17];
  const float* pwf_b  = (const float*)d_in[18];
  const float* lpi_w  = (const float*)d_in[19];
  const float* lpi_b  = (const float*)d_in[20];
  const float* lni_g  = (const float*)d_in[21];
  const float* lni_b  = (const float*)d_in[22];
  const float* lnf_g  = (const float*)d_in[23];
  const float* lnf_b  = (const float*)d_in[24];
  const float* mlpf_w = (const float*)d_in[25];
  const float* mlpf_b = (const float*)d_in[26];

  char* ws = (char*)d_ws;
  size_t off = 0;
  auto alloc = [&](size_t bytes) -> void* {
    void* p = ws + off;
    off += (bytes + 255) & ~(size_t)255;
    return p;
  };
  // Aliased lifetimes: BF_X = inp_bf16 -> mha_bf16 -> res_inp_bf16
  //                    BF_T = t_bf16 -> res_bf16 ; BF_H = h1 -> r1
  //                    F32A = v -> mha(f32) -> res_inp(f32)
  unsigned short* BF_X  = (unsigned short*)alloc((size_t)Nrows * Dd * 2);
  unsigned short* BF_T  = (unsigned short*)alloc((size_t)Nrows * Dd * 2);
  unsigned short* BF_H  = (unsigned short*)alloc((size_t)Nrows * DFFc * 2);
  float*          F32A  = (float*)alloc((size_t)Nrows * Dd * 4);
  unsigned short* CTX2  = (unsigned short*)alloc((size_t)Bb * 2 * Dd * 2);
  float*          G2    = (float*)alloc((size_t)Bb * 2 * Dd * 4);
  unsigned short* Wwv   = (unsigned short*)alloc((size_t)Dd * Dd * 2);
  unsigned short* Wfc   = (unsigned short*)alloc((size_t)Dd * Dd * 2);
  unsigned short* Wlpc  = (unsigned short*)alloc((size_t)Dd * Dd * 2);
  unsigned short* W1b   = (unsigned short*)alloc((size_t)DFFc * Dd * 2);
  unsigned short* W2b   = (unsigned short*)alloc((size_t)Dd * DFFc * 2);
  unsigned short* Wpwf  = (unsigned short*)alloc((size_t)DFFc * Dd * 2);
  unsigned short* Wlpi  = (unsigned short*)alloc((size_t)Dd * DFFc * 2);
  unsigned short* Wmlpf = (unsigned short*)alloc((size_t)240 * Dd * 2);

  auto cvt = [&](const float* s, unsigned short* d, int n) {
    k_cvt_bf16<<<(n + 255) / 256, 256, 0, stream>>>(s, d, n);
  };
  cvt(inp, BF_X, Nrows * Dd);
  cvt(wv, Wwv, Dd * Dd);   cvt(fc_w, Wfc, Dd * Dd);   cvt(lpc_w, Wlpc, Dd * Dd);
  cvt(w1, W1b, DFFc * Dd); cvt(w2, W2b, Dd * DFFc);
  cvt(pwf_w, Wpwf, DFFc * Dd); cvt(lpi_w, Wlpi, Dd * DFFc);
  k_pad_mlpf<<<(240 * Dd + 255) / 256, 256, 0, stream>>>(mlpf_w, Wmlpf);

  const int MB = Nrows / 16;   // 3200 row-tiles

  // v = inp @ wv^T + bv                                   (f32 -> F32A)
  k_gemm<512, 32, 0, 0, 1><<<MB, BT, 0, stream>>>(BF_X, Wwv, bv, F32A, Dd);
  // ctx closed form -> 2 distinct rows/batch              (bf16 -> CTX2)
  k_ctx<<<Bb, BT, 0, stream>>>(F32A, mask, CTX2);
  // g2 = ctx2rows @ fc_w^T  (only 1024 rows!)             (f32 -> G2)
  k_gemm<512, 32, 0, 0, 0><<<(Bb * 2) / 16, BT, 0, stream>>>(CTX2, Wfc, nullptr, G2, Dd);
  // mha = LN_torch(select + fc_b + inp)                   (F32A + BF_X)
  k_mha_ln<<<Nrows, BT, 0, stream>>>(G2, mask, fc_b, inp, lnm_g, lnm_b, F32A, BF_X);
  // h1 = relu(mha @ w1^T + b1)                            (bf16 -> BF_H)
  k_gemm<512, 16, 1, 1, 1><<<MB, BT, 0, stream>>>(BF_X, W1b, b1, BF_H, DFFc);
  // ctx2 = LN_c(h1 @ w2^T + b2 + mha); t = ctx2 + inp     (bf16 -> BF_T)
  k_gemm_ln<256, 0><<<MB, BT, 0, stream>>>(BF_H, W2b, b2, 1.0f, F32A,
                                           lni_g, lni_b, inp, nullptr, BF_T);
  // res_inp = LN_c(t @ lpc_w^T + 2*lpc_b)                 (F32A + BF_X)
  k_gemm_ln<512, 1><<<MB, BT, 0, stream>>>(BF_T, Wlpc, lpc_b, 2.0f, nullptr,
                                           lnc_g, lnc_b, nullptr, F32A, BF_X);
  // r1 = relu(res_inp @ pwf_w^T + pwf_b)                  (bf16 -> BF_H)
  k_gemm<512, 16, 1, 1, 1><<<MB, BT, 0, stream>>>(BF_X, Wpwf, pwf_b, BF_H, DFFc);
  // res = LN_c(r1 @ lpi_w^T + lpi_b + res_inp)            (bf16 -> BF_T)
  k_gemm_ln<256, 2><<<MB, BT, 0, stream>>>(BF_H, Wlpi, lpi_b, 1.0f, F32A,
                                           lnf_g, lnf_b, nullptr, nullptr, BF_T);
  // out = res @ mlpf_w^T + mlpf_b  (230 cols)             (f32 -> d_out)
  k_gemm<512, 15, 0, 0, 1><<<MB, BT, 0, stream>>>(BF_T, Wmlpf, mlpf_b, d_out, HIDc);
}
